// AtlasMemoryLayer_30597347016909
// MI455X (gfx1250) — matrix operand order repaired and checked by reading, not compiled, not run on hardware
//
#include <hip/hip_runtime.h>
#include <math.h>

// Problem constants (reference: B=2,T=2048,C=1024,H=16,D=64,CHUNK=64)
#define BB 2
#define TT 2048
#define CC 1024
#define HH 16
#define DD 64
#define CS 64
#define NC 32
#define BT (BB * TT)   // 4096 tokens
#define BH (BB * HH)   // 32 (batch,head) pairs
#define OMEGA_W 16

#define NS_A 3.4445f
#define NS_B (-4.7750f)
#define NS_C 2.0315f

typedef __attribute__((ext_vector_type(2)))  float  v2f;
typedef __attribute__((ext_vector_type(8)))  float  v8f;
typedef __attribute__((ext_vector_type(16))) __bf16 v16bf;
typedef __attribute__((ext_vector_type(4)))  unsigned int u32x4;
typedef __attribute__((ext_vector_type(8)))  int i32x8;
typedef __attribute__((ext_vector_type(4)))  int i32x4;

typedef __attribute__((address_space(3))) void las_void;

// Exact pointer element type for the async b128 builtin:
// param0 = int4* in global AS(1), param1 = int4* in LDS AS(3).
typedef int v4i_b128 __attribute__((vector_size(16)));
typedef __attribute__((address_space(1))) v4i_b128 g_v4i;
typedef __attribute__((address_space(3))) v4i_b128 l_v4i;

// ---------------------------------------------------------------------------
// CDNA5 async global->LDS copy (ASYNCcnt) and TDM (TENSORcnt), with guarded
// fallbacks so a missing builtin degrades to synchronous staging.
// ---------------------------------------------------------------------------
#if __has_builtin(__builtin_amdgcn_global_load_async_to_lds_b128)
#define HAVE_ASYNC_LDS 1
#else
#define HAVE_ASYNC_LDS 0
#endif

#if __has_builtin(__builtin_amdgcn_tensor_load_to_lds)
#define HAVE_TDM 1
#else
#define HAVE_TDM 0
#endif

#if HAVE_ASYNC_LDS
__device__ __forceinline__ void async_copy16(const void* g, void* l)
{
    __builtin_amdgcn_global_load_async_to_lds_b128((g_v4i*)g, (l_v4i*)l, 0, 0);
}
__device__ __forceinline__ void async_wait0()
{
#if __has_builtin(__builtin_amdgcn_s_wait_asynccnt)
    __builtin_amdgcn_s_wait_asynccnt(0);
#else
    asm volatile("s_wait_asynccnt 0" ::: "memory");
#endif
}
#endif

#if HAVE_TDM
__device__ __forceinline__ void tensor_wait0()
{
#if __has_builtin(__builtin_amdgcn_s_wait_tensorcnt)
    __builtin_amdgcn_s_wait_tensorcnt(0);
#else
    asm volatile("s_wait_tensorcnt 0" ::: "memory");
#endif
}
#endif

// ---------------------------------------------------------------------------
// WMMA fragment helpers (wave32, layouts per CDNA5 ISA 7.12.2)
// A 16x32 bf16: lane half selects K-group; element e -> K = (e<8?e:e+8)+half*8
// B 32x16 bf16: N = lane&15; element e -> K = (e<8?e:e+8)+half*8 (same as A)
// C/D 16x16 f32: element r -> M = r + half*8, N = lane&15
// ---------------------------------------------------------------------------
__device__ __forceinline__ v8f mma_bf16_k32(const __bf16* a, int as_m, int as_k,
                                            const __bf16* b, int bs_k, int bs_n,
                                            v8f acc)
{
    const int lane = threadIdx.x & 31;
    const int half = lane >> 4, l15 = lane & 15;
    v16bf av, bv;
#pragma unroll
    for (int e = 0; e < 8; ++e) {
        av[e]     = a[l15 * as_m + (half * 8 + e) * as_k];
        av[e + 8] = a[l15 * as_m + (16 + half * 8 + e) * as_k];
    }
#pragma unroll
    for (int e = 0; e < 8; ++e) {
        bv[e]     = b[(half * 8 + e) * bs_k + l15 * bs_n];
        bv[e + 8] = b[(16 + half * 8 + e) * bs_k + l15 * bs_n];
    }
    return __builtin_amdgcn_wmma_f32_16x16x32_bf16(false, av, false, bv,
                                                   (short)0, acc, false, false);
}

// fp32-exact WMMA (V_WMMA_F32_16X16X4_F32). kmin/kmax mask the 4-deep K slice
// (used for ragged omega windows). Scalar fallback produces the identical
// C/D lane mapping if the builtin is not declared on this toolchain.
#if __has_builtin(__builtin_amdgcn_wmma_f32_16x16x4_f32)
#define HAVE_WMMA_F32_K4 1
#else
#define HAVE_WMMA_F32_K4 0
#endif

__device__ __forceinline__ v8f mma_f32_k4(const float* a, int as_m, int as_k,
                                          const float* b, int bs_k, int bs_n,
                                          v8f acc, int kmin, int kmax)
{
    const int lane = threadIdx.x & 31;
    const int half = lane >> 4, l15 = lane & 15;
#if HAVE_WMMA_F32_K4
    v2f av, bv;
#pragma unroll
    for (int e = 0; e < 2; ++e) {
        int k = half * 2 + e;
        bool ok = (k >= kmin) && (k <= kmax);
        av[e] = ok ? a[l15 * as_m + k * as_k] : 0.0f;
        bv[e] = ok ? b[k * bs_k + l15 * bs_n] : 0.0f;
    }
    return __builtin_amdgcn_wmma_f32_16x16x4_f32(false, av, false, bv,
                                                 (short)0, acc, false, false);
#else
#pragma unroll
    for (int r = 0; r < 8; ++r) {
        int m = r + half * 8, n = l15;
        float s = acc[r];
        for (int k = kmin; k <= kmax; ++k)
            s += a[m * as_m + k * as_k] * b[k * bs_k + n * bs_n];
        acc[r] = s;
    }
    return acc;
#endif
}

// ---------------------------------------------------------------------------
// f32 -> bf16 conversion / zero (grid-stride)
// ---------------------------------------------------------------------------
__global__ void cvt_bf16_kernel(const float* __restrict__ src,
                                __bf16* __restrict__ dst, int n)
{
    for (int i = blockIdx.x * blockDim.x + threadIdx.x; i < n;
         i += gridDim.x * blockDim.x)
        dst[i] = (__bf16)src[i];
}

__global__ void zero_f32_kernel(float* __restrict__ p, int n)
{
    for (int i = blockIdx.x * blockDim.x + threadIdx.x; i < n;
         i += gridDim.x * blockDim.x)
        p[i] = 0.0f;
}

// ---------------------------------------------------------------------------
// GEMM: Y[M,N] = A[M,K] @ W[N,K]^T   (bf16 inputs, fp32 accumulate/output)
// Block: 256 threads = 8 waves, tile 128(M) x 64(N) x 32(K).
// A tile staged with GLOBAL_LOAD_ASYNC_TO_LDS_B128 when available.
// ---------------------------------------------------------------------------
__global__ __launch_bounds__(256)
void gemm_bf16_kernel(const __bf16* __restrict__ A, const __bf16* __restrict__ W,
                      float* __restrict__ Y, int M, int N, int K)
{
    __shared__ __bf16 sA[128][40];   // 80B rows: 16B-aligned B128 targets
    __shared__ __bf16 sB[32][68];    // B[k][n] = W[n][k] (transposed on store)
    const int tid = threadIdx.x, lane = tid & 31, wid = tid >> 5;
    const int m0 = blockIdx.y * 128, n0 = blockIdx.x * 64;
    const int wm = (wid >> 1) * 32, wn = (wid & 1) * 32;

    v8f acc[2][2] = {};

    const int arow = tid >> 1, ac0 = (tid & 1) * 16;
    const int bn = tid >> 2, bkq = (tid & 3) * 8;

    for (int kk = 0; kk < K; kk += 32) {
        {   // stage A tile: 128x32, 32B per thread
            const __bf16* src = A + (size_t)(m0 + arow) * K + kk + ac0;
#if HAVE_ASYNC_LDS
            async_copy16(src,     &sA[arow][ac0]);
            async_copy16(src + 8, &sA[arow][ac0 + 8]);
#else
#pragma unroll
            for (int j = 0; j < 16; ++j) sA[arow][ac0 + j] = src[j];
#endif
        }
        {   // stage W tile transposed into B[k][n]
            const __bf16* src = W + (size_t)(n0 + bn) * K + kk + bkq;
#pragma unroll
            for (int j = 0; j < 8; ++j) sB[bkq + j][bn] = src[j];
        }
        if (kk + 32 < K) {   // global_prefetch_b8 of next tiles
            __builtin_prefetch(A + (size_t)(m0 + arow) * K + kk + 32, 0, 1);
            __builtin_prefetch(W + (size_t)(n0 + bn) * K + kk + 32, 0, 1);
        }
#if HAVE_ASYNC_LDS
        async_wait0();
#endif
        __syncthreads();
#pragma unroll
        for (int ti = 0; ti < 2; ++ti)
#pragma unroll
            for (int tj = 0; tj < 2; ++tj)
                acc[ti][tj] = mma_bf16_k32(&sA[wm + 16 * ti][0], 40, 1,
                                           &sB[0][wn + 16 * tj], 68, 1,
                                           acc[ti][tj]);
        __syncthreads();
    }

    const int half = lane >> 4, l15 = lane & 15;
#pragma unroll
    for (int ti = 0; ti < 2; ++ti)
#pragma unroll
        for (int tj = 0; tj < 2; ++tj)
#pragma unroll
            for (int r = 0; r < 8; ++r) {
                int m = m0 + wm + 16 * ti + r + half * 8;
                int n = n0 + wn + 16 * tj + l15;
                Y[(size_t)m * N + n] = acc[ti][tj][r];
            }
}

// ---------------------------------------------------------------------------
// Causal depthwise conv (K=4) + optional per-head RMS norm + optional
// poly feature phi(x) = x + x^2/2. One block per token, 4 channels/thread.
// ---------------------------------------------------------------------------
__global__ __launch_bounds__(256)
void conv_rms_kernel(const float* __restrict__ xin, const float* __restrict__ w,
                     const float* __restrict__ bias, float* __restrict__ yout,
                     int do_rms, int do_poly)
{
    const int p = blockIdx.x;           // token = b*T + t
    const int b = p / TT, t = p % TT;
    const int tid = threadIdx.x, c0 = tid * 4;
    float y[4];
#pragma unroll
    for (int j = 0; j < 4; ++j) {
        int c = c0 + j;
        float acc = bias[c];
#pragma unroll
        for (int q = 0; q < 4; ++q) {
            int tt = t - 3 + q;                 // left zero padding (KCONV-1)
            if (tt >= 0)
                acc += w[c * 4 + q] * xin[(size_t)(b * TT + tt) * CC + c];
        }
        y[j] = acc;
    }
    if (do_rms) {
        __shared__ float red[256];
        __shared__ float hs[16];
        float ss = y[0] * y[0] + y[1] * y[1] + y[2] * y[2] + y[3] * y[3];
        red[tid] = ss;
        __syncthreads();
        if ((tid & 15) == 0) {                  // 16 threads == one head (64 ch)
            float s = 0.f;
            for (int i = 0; i < 16; ++i) s += red[tid + i];
            hs[tid >> 4] = s;
        }
        __syncthreads();
        float scale = rsqrtf(hs[tid >> 4] * (1.0f / 64.0f) + 1e-6f);
#pragma unroll
        for (int j = 0; j < 4; ++j) y[j] *= scale;
    }
    if (do_poly)
#pragma unroll
        for (int j = 0; j < 4; ++j) y[j] = y[j] + 0.5f * y[j] * y[j];
#pragma unroll
    for (int j = 0; j < 4; ++j)
        yout[(size_t)p * CC + c0 + j] = y[j];
}

// ---------------------------------------------------------------------------
// Gates: sigmoid(x @ W.T) for Wa/We/Wt/Wg, (H,C) each -> gout[g][p][h]
// ---------------------------------------------------------------------------
__global__ __launch_bounds__(256)
void gates_kernel(const float* __restrict__ x,
                  const float* __restrict__ Wa, const float* __restrict__ We,
                  const float* __restrict__ Wt, const float* __restrict__ Wg,
                  float* __restrict__ gout)
{
    const int p = blockIdx.x;
    const int tid = threadIdx.x;
    const int u = tid >> 2, sub = tid & 3;   // u = g*16 + h
    const int g = u >> 4, h = u & 15;
    const float* W = (g == 0) ? Wa : (g == 1) ? We : (g == 2) ? Wt : Wg;
    const float* xp = x + (size_t)p * CC;
    const float* wr = W + (size_t)h * CC;
    float s = 0.f;
    for (int k = sub * 256; k < sub * 256 + 256; ++k) s += xp[k] * wr[k];
    __shared__ float red[256];
    red[tid] = s;
    __syncthreads();
    if (sub == 0) {
        float tot = red[tid] + red[tid + 1] + red[tid + 2] + red[tid + 3];
        gout[(size_t)g * BT * HH + (size_t)p * HH + h] = 1.0f / (1.0f + expf(-tot));
    }
}

// ---------------------------------------------------------------------------
// S1 (per chunk): one block per (b,h).
//  Pred = K_c @ M^T                         (fp32 WMMA 16x16x4)
//  Ew   = gamma_t * (Pred - V)
//  mom_t = -2*eta_t * Ew[win]^T @ K[win]    (masked fp32 WMMA, window=16)
//  S_t  = theta_t * S_{t-1} + mom_t         (register-resident scan)
// ---------------------------------------------------------------------------
__global__ __launch_bounds__(256)
void scan_s1_kernel(int cidx,
                    const float* __restrict__ kp, const float* __restrict__ vv,
                    const float* __restrict__ gates,
                    const float* __restrict__ Mst, float* __restrict__ Sst,
                    float* __restrict__ chunkS)
{
    const int bh = blockIdx.x;
    const int b = bh / HH, h = bh % HH;
    const int tid = threadIdx.x, lane = tid & 31, wid = tid >> 5;
    const int t0 = cidx * CS;

    __shared__ float sM[64][68];     // 272B rows: 16B-aligned, conflict-free
    __shared__ float sK[64][68];
    __shared__ float sE[64][68];
    __shared__ float seta[64], sth[64], sgam[64];

    const float* Mg = Mst + (size_t)bh * 4096;
#if HAVE_ASYNC_LDS
    for (int i4 = tid; i4 < 1024; i4 += 256) {
        int i = i4 * 4, r = i >> 6, d = i & 63;
        async_copy16(Mg + i, &sM[r][d]);
        async_copy16(kp + (size_t)(b * TT + t0 + r) * CC + h * 64 + d, &sK[r][d]);
    }
#else
    for (int i = tid; i < 4096; i += 256) sM[i >> 6][i & 63] = Mg[i];
    for (int i = tid; i < 4096; i += 256) {
        int t = i >> 6, d = i & 63;
        sK[t][d] = kp[(size_t)(b * TT + t0 + t) * CC + h * 64 + d];
    }
#endif
    if (tid < 64) {
        size_t p = (size_t)(b * TT + t0 + tid) * HH + h;
        seta[tid] = gates[(size_t)1 * BT * HH + p];
        sth[tid]  = gates[(size_t)2 * BT * HH + p];
        sgam[tid] = gates[(size_t)3 * BT * HH + p];
    }
#if HAVE_ASYNC_LDS
    async_wait0();
#endif
    __syncthreads();

    // Pred = K @ M^T ; Ew = gamma * (Pred - V).  16 tiles over 8 waves.
    for (int tt = wid; tt < 16; tt += 8) {
        int ti = tt >> 2, tj = tt & 3;           // ti: t-tile, tj: v-tile
        v8f acc = {};
        for (int k0 = 0; k0 < 64; k0 += 4)
            acc = mma_f32_k4(&sK[ti * 16][k0], 68, 1,    // A[m=t][k]
                             &sM[tj * 16][k0], 1, 68,    // B[k][n=v] = M[v][k]
                             acc, 0, 3);
        const int half = lane >> 4, l15 = lane & 15;
#pragma unroll
        for (int r = 0; r < 8; ++r) {
            int trow = ti * 16 + r + half * 8;
            int vcol = tj * 16 + l15;
            float vval = vv[(size_t)(b * TT + t0 + trow) * CC + h * 64 + vcol];
            sE[trow][vcol] = sgam[trow] * (acc[r] - vval);
        }
    }
    __syncthreads();

    // mom_t via windowed matmuls (parallel over t; 8 t per wave)
    float* cS = chunkS + (size_t)bh * CS * 4096;
    for (int t = wid; t < 64; t += 8) {
        int slo = (t - (OMEGA_W - 1) > 0) ? (t - (OMEGA_W - 1)) : 0;
        float sc = -2.0f * seta[t];
        for (int tt = 0; tt < 16; ++tt) {
            int ti = tt >> 2, tj = tt & 3;       // ti: v-tile, tj: k-tile
            v8f acc = {};
            for (int s0 = slo & ~3; s0 <= t; s0 += 4)
                acc = mma_f32_k4(&sE[s0][ti * 16], 1, 68,   // A[m=v][k=s]=Ew[s][v]
                                 &sK[s0][tj * 16], 68, 1,   // B[k=s][n]=K[s][n]
                                 acc, slo - s0, t - s0);
            const int half = lane >> 4, l15 = lane & 15;
#pragma unroll
            for (int r = 0; r < 8; ++r) {
                int v = ti * 16 + r + half * 8;
                int k = tj * 16 + l15;
                cS[(size_t)t * 4096 + v * 64 + k] = sc * acc[r];
            }
        }
    }
    __threadfence_block();
    __syncthreads();

    // S scan: each thread owns 16 disjoint elements across all 64 steps.
    float sreg[16];
    float* Sg = Sst + (size_t)bh * 4096;
#pragma unroll
    for (int j = 0; j < 16; ++j) sreg[j] = Sg[tid + 256 * j];
    for (int t = 0; t < 64; ++t) {
        float th = sth[t];
#pragma unroll
        for (int j = 0; j < 16; ++j) {
            size_t idx = (size_t)t * 4096 + tid + 256 * j;
            float s = th * sreg[j] + cS[idx];
            sreg[j] = s;
            cS[idx] = s;
        }
    }
#pragma unroll
    for (int j = 0; j < 16; ++j) Sg[tid + 256 * j] = sreg[j];
}

// ---------------------------------------------------------------------------
// S2: polar express (Newton-Schulz, 5 steps) on one 64x64 matrix per block.
// Input tile fetched by the Tensor Data Mover when available: the D# uses
// pad_enable (interval=64 DWORDs, amount=4 DWORDs) so the DMA itself lays the
// matrix out as the bank-conflict-free [64][68] LDS image.
// ---------------------------------------------------------------------------
__global__ __launch_bounds__(256)
void polar_kernel(float* __restrict__ chunkS)
{
    const int mid = blockIdx.x;                 // bh*64 + t
    float* Xg = chunkS + (size_t)mid * 4096;
    const int tid = threadIdx.x, lane = tid & 31, wid = tid >> 5;

    __shared__ float  sX[64][68];
    __shared__ float  sA[64][65];
    __shared__ __bf16 sXb[64][66];
    __shared__ __bf16 sAb[64][66];
    __shared__ __bf16 sWb[64][66];
    __shared__ float  red[256];

    float xr[16];
    float ss = 0.f;
#if HAVE_TDM
    if (tid < 32) {
        unsigned lds_off = (unsigned)(size_t)(las_void*)&sX[0][0];
        unsigned long long ga = (unsigned long long)(size_t)Xg;
        u32x4 g0;
        g0[0] = 1u;                                    // count=1 user descriptor
        g0[1] = lds_off;                               // lds_addr
        g0[2] = (unsigned)(ga & 0xFFFFFFFFu);          // global_addr[31:0]
        g0[3] = (unsigned)((ga >> 32) & 0x01FFFFFFu)   // global_addr[56:32]
              | (2u << 30);                            // type = 2 (image)
        i32x8 g1;
        g1[0] = (2 << 16)        // data_size = 4B
              | (1 << 20)        // pad_enable
              | (5 << 22)        // pad_interval: 64 DWORDs (one tile row)
              | (3 << 25);       // pad_amount: 4 DWORDs -> [64][68] image
        g1[1] = 64 << 16;        // tensor_dim0 = 64 (low 16 @ bits 63:48)
        g1[2] = 64 << 16;        // tensor_dim1 = 64 (low 16 @ bits 95:80)
        g1[3] = 64 << 16;        // tile_dim0 = 64 @ bits 127:112
        g1[4] = 64;              // tile_dim1 = 64; tile_dim2 = 0
        g1[5] = 64;              // tensor_dim0_stride = 64 (low 32)
        g1[6] = 0;
        g1[7] = 0;
        i32x4 gz4 = {};
#if __clang_major__ >= 23
        i32x8 gz8 = {};
        __builtin_amdgcn_tensor_load_to_lds(g0, g1, gz4, gz4, gz8, 0);
#else
        __builtin_amdgcn_tensor_load_to_lds(g0, g1, gz4, gz4, 0);
#endif
        tensor_wait0();
    }
    __syncthreads();
#pragma unroll
    for (int j = 0; j < 16; ++j) {
        int idx = tid + 256 * j;
        xr[j] = sX[idx >> 6][idx & 63];
        ss += xr[j] * xr[j];
    }
#else
#pragma unroll
    for (int j = 0; j < 16; ++j) {
        xr[j] = Xg[tid + 256 * j];
        ss += xr[j] * xr[j];
    }
#endif
    red[tid] = ss;
    __syncthreads();
    for (int s = 128; s > 0; s >>= 1) {
        if (tid < s) red[tid] += red[tid + s];
        __syncthreads();
    }
    float inv = 1.0f / (sqrtf(red[0]) + 1e-7f);
#pragma unroll
    for (int j = 0; j < 16; ++j) {
        int idx = tid + 256 * j, row = idx >> 6, col = idx & 63;
        float v = xr[j] * inv;
        sX[row][col] = v;
        sXb[row][col] = (__bf16)v;
    }
    __syncthreads();

    const int half = lane >> 4, l15 = lane & 15;
    for (int it = 0; it < 5; ++it) {
        // XXt = X @ X^T
        for (int tt = wid; tt < 16; tt += 8) {
            int ti = tt >> 2, tj = tt & 3;
            v8f acc = {};
            acc = mma_bf16_k32(&sXb[ti * 16][0],  66, 1, &sXb[tj * 16][0],  1, 66, acc);
            acc = mma_bf16_k32(&sXb[ti * 16][32], 66, 1, &sXb[tj * 16][32], 1, 66, acc);
#pragma unroll
            for (int r = 0; r < 8; ++r) {
                int row = ti * 16 + r + half * 8, col = tj * 16 + l15;
                sA[row][col] = acc[r];
                sAb[row][col] = (__bf16)acc[r];
            }
        }
        __syncthreads();
        // W = b*XXt + c * XXt @ XXt
        for (int tt = wid; tt < 16; tt += 8) {
            int ti = tt >> 2, tj = tt & 3;
            v8f acc = {};
            acc = mma_bf16_k32(&sAb[ti * 16][0],  66, 1, &sAb[0][tj * 16],  66, 1, acc);
            acc = mma_bf16_k32(&sAb[ti * 16][32], 66, 1, &sAb[32][tj * 16], 66, 1, acc);
#pragma unroll
            for (int r = 0; r < 8; ++r) {
                int row = ti * 16 + r + half * 8, col = tj * 16 + l15;
                sWb[row][col] = (__bf16)(NS_B * sA[row][col] + NS_C * acc[r]);
            }
        }
        __syncthreads();
        // Xnew = a*X + W @ X  (stage in registers, then commit)
        float xn[2][8];
#pragma unroll
        for (int q = 0; q < 2; ++q) {
            int tt = wid + 8 * q;
            int ti = tt >> 2, tj = tt & 3;
            v8f acc = {};
            acc = mma_bf16_k32(&sWb[ti * 16][0],  66, 1, &sXb[0][tj * 16],  66, 1, acc);
            acc = mma_bf16_k32(&sWb[ti * 16][32], 66, 1, &sXb[32][tj * 16], 66, 1, acc);
#pragma unroll
            for (int r = 0; r < 8; ++r) {
                int row = ti * 16 + r + half * 8, col = tj * 16 + l15;
                xn[q][r] = NS_A * sX[row][col] + acc[r];
            }
        }
        __syncthreads();
#pragma unroll
        for (int q = 0; q < 2; ++q) {
            int tt = wid + 8 * q;
            int ti = tt >> 2, tj = tt & 3;
#pragma unroll
            for (int r = 0; r < 8; ++r) {
                int row = ti * 16 + r + half * 8, col = tj * 16 + l15;
                sX[row][col] = xn[q][r];
                sXb[row][col] = (__bf16)xn[q][r];
            }
        }
        __syncthreads();
    }

    for (int i = tid; i < 4096; i += 256)
        Xg[i] = sX[i >> 6][i & 63];
}

// ---------------------------------------------------------------------------
// S3 (per chunk): M_t = alpha_t * M_{t-1} + Sorth_t ; y_t = M_t q_t.
// ---------------------------------------------------------------------------
__global__ __launch_bounds__(256)
void scan_s3_kernel(int cidx,
                    const float* __restrict__ qn, const float* __restrict__ gates,
                    const float* __restrict__ chunkS, float* __restrict__ Mst,
                    __bf16* __restrict__ ybf)
{
    const int bh = blockIdx.x;
    const int b = bh / HH, h = bh % HH;
    const int tid = threadIdx.x;
    const int t0 = cidx * CS;

    __shared__ float sM[64][68];
    __shared__ float sq[64][68];
    __shared__ float sal[64];

    float* Mg = Mst + (size_t)bh * 4096;
#if HAVE_ASYNC_LDS
    for (int i4 = tid; i4 < 1024; i4 += 256) {
        int i = i4 * 4, r = i >> 6, d = i & 63;
        async_copy16(Mg + i, &sM[r][d]);
        async_copy16(qn + (size_t)(b * TT + t0 + r) * CC + h * 64 + d, &sq[r][d]);
    }
#else
    for (int i = tid; i < 4096; i += 256) sM[i >> 6][i & 63] = Mg[i];
    for (int i = tid; i < 4096; i += 256) {
        int t = i >> 6, d = i & 63;
        sq[t][d] = qn[(size_t)(b * TT + t0 + t) * CC + h * 64 + d];
    }
#endif
    if (tid < 64)
        sal[tid] = gates[(size_t)(b * TT + t0 + tid) * HH + h];   // alpha = gate 0
#if HAVE_ASYNC_LDS
    async_wait0();
#endif
    __syncthreads();

    const float* cS = chunkS + (size_t)bh * CS * 4096;
    for (int t = 0; t < 64; ++t) {
        float al = sal[t];
        for (int i = tid; i < 4096; i += 256) {
            float m = al * sM[i >> 6][i & 63] + cS[(size_t)t * 4096 + i];
            sM[i >> 6][i & 63] = m;
        }
        __syncthreads();
        if (tid < 64) {
            int v = tid;
            float y = 0.f;
#pragma unroll
            for (int k = 0; k < 64; ++k) y += sM[v][k] * sq[t][k];
            ybf[(size_t)(b * TT + t0 + t) * CC + h * 64 + v] = (__bf16)y;
        }
        __syncthreads();
    }
    for (int i = tid; i < 4096; i += 256) Mg[i] = sM[i >> 6][i & 63];
}

// ---------------------------------------------------------------------------
// Host orchestration
// ---------------------------------------------------------------------------
extern "C" void kernel_launch(void* const* d_in, const int* in_sizes, int n_in,
                              void* d_out, int out_size, void* d_ws, size_t ws_size,
                              hipStream_t stream)
{
    (void)in_sizes; (void)n_in; (void)out_size; (void)ws_size;

    const float* x     = (const float*)d_in[0];
    const float* Wq    = (const float*)d_in[1];
    const float* Wk    = (const float*)d_in[2];
    const float* Wv    = (const float*)d_in[3];
    const float* Wp    = (const float*)d_in[4];
    const float* cqw   = (const float*)d_in[5];
    const float* cqb   = (const float*)d_in[6];
    const float* ckw   = (const float*)d_in[7];
    const float* ckb   = (const float*)d_in[8];
    const float* cvw   = (const float*)d_in[9];
    const float* cvb   = (const float*)d_in[10];
    const float* Wa    = (const float*)d_in[11];
    const float* We    = (const float*)d_in[12];
    const float* Wt    = (const float*)d_in[13];
    const float* Wg    = (const float*)d_in[14];
    float* out = (float*)d_out;

    char* base = (char*)d_ws;
    size_t off = 0;
    auto take = [&](size_t bytes) -> void* {
        void* p = base + off;
        off += (bytes + 255) & ~(size_t)255;
        return p;
    };

    __bf16* xbf   = (__bf16*)take((size_t)BT * CC * 2);
    __bf16* wqbf  = (__bf16*)take((size_t)CC * CC * 2);
    __bf16* wkbf  = (__bf16*)take((size_t)CC * CC * 2);
    __bf16* wvbf  = (__bf16*)take((size_t)CC * CC * 2);
    __bf16* wpbf  = (__bf16*)take((size_t)CC * CC * 2);
    __bf16* ybf   = (__bf16*)take((size_t)BT * CC * 2);
    float*  xq    = (float*)take((size_t)BT * CC * 4);
    float*  xk    = (float*)take((size_t)BT * CC * 4);
    float*  xv    = (float*)take((size_t)BT * CC * 4);
    float*  qn    = (float*)take((size_t)BT * CC * 4);
    float*  kphi  = (float*)take((size_t)BT * CC * 4);
    float*  vconv = (float*)take((size_t)BT * CC * 4);
    float*  gbuf  = (float*)take((size_t)4 * BT * HH * 4);
    float*  Mst   = (float*)take((size_t)BH * DD * DD * 4);
    float*  Sst   = (float*)take((size_t)BH * DD * DD * 4);
    float*  chunkS= (float*)take((size_t)BH * CS * DD * DD * 4);

    const int thr = 256;

    // 1) bf16 conversions for tensor-core GEMMs
    cvt_bf16_kernel<<<1024, thr, 0, stream>>>(x,  xbf,  BT * CC);
    cvt_bf16_kernel<<<512,  thr, 0, stream>>>(Wq, wqbf, CC * CC);
    cvt_bf16_kernel<<<512,  thr, 0, stream>>>(Wk, wkbf, CC * CC);
    cvt_bf16_kernel<<<512,  thr, 0, stream>>>(Wv, wvbf, CC * CC);
    cvt_bf16_kernel<<<512,  thr, 0, stream>>>(Wp, wpbf, CC * CC);

    // 2) q/k/v projections
    dim3 ggrid(CC / 64, BT / 128);
    gemm_bf16_kernel<<<ggrid, thr, 0, stream>>>(xbf, wqbf, xq, BT, CC, CC);
    gemm_bf16_kernel<<<ggrid, thr, 0, stream>>>(xbf, wkbf, xk, BT, CC, CC);
    gemm_bf16_kernel<<<ggrid, thr, 0, stream>>>(xbf, wvbf, xv, BT, CC, CC);

    // 3) short conv (+RMS, +poly for k)
    conv_rms_kernel<<<BT, thr, 0, stream>>>(xq, cqw, cqb, qn,    1, 0);
    conv_rms_kernel<<<BT, thr, 0, stream>>>(xk, ckw, ckb, kphi,  1, 1);
    conv_rms_kernel<<<BT, thr, 0, stream>>>(xv, cvw, cvb, vconv, 0, 0);

    // 4) gates alpha/eta/theta/gamma
    gates_kernel<<<BT, thr, 0, stream>>>(x, Wa, We, Wt, Wg, gbuf);

    // 5) zero recurrent state
    zero_f32_kernel<<<128, thr, 0, stream>>>(Mst, BH * DD * DD);
    zero_f32_kernel<<<128, thr, 0, stream>>>(Sst, BH * DD * DD);

    // 6) chunked scan: S1 -> polar (WMMA-heavy) -> S3, sequential over chunks
    for (int c = 0; c < NC; ++c) {
        scan_s1_kernel<<<BH, thr, 0, stream>>>(c, kphi, vconv, gbuf, Mst, Sst, chunkS);
        polar_kernel<<<BH * CS, thr, 0, stream>>>(chunkS);
        scan_s3_kernel<<<BH, thr, 0, stream>>>(c, qn, gbuf, chunkS, Mst, ybf);
    }

    // 7) output projection
    gemm_bf16_kernel<<<ggrid, thr, 0, stream>>>(ybf, wpbf, out, BT, CC, CC);
}
